// STGRF_GRU_24275155157315
// MI455X (gfx1250) — matrix operand order repaired and checked by reading, not compiled, not run on hardware
//
#include <hip/hip_runtime.h>
#include <hip/hip_bf16.h>

typedef _Float16 f16;
typedef __attribute__((ext_vector_type(16))) _Float16 v16h;
typedef __attribute__((ext_vector_type(4)))  _Float16 h4;
typedef __attribute__((ext_vector_type(8)))  float    v8f;
typedef __attribute__((ext_vector_type(4)))  float    f4v;

#define B_      4
#define HIST_   24
#define PRED_   12
#define N_      10000
#define E_      160000
#define IN_DIM_ 32
#define FEAT_   31
#define E_H_    32
#define E_OUT_  30
#define GNN_OUT_ 9
#define HID_    64
#define K1_PAD  96   // 67 -> 3 chunks of 32
#define KXC_PAD 64   // 41 -> 2 chunks of 32
#define NG_     192  // 3*HID
#define EDGE_TPW 5   // edge tiles per wave

// fast activations: v_rcp_f32 + v_exp_f32 instead of IEEE div / libm tanh
__device__ __forceinline__ float sigm(float x) {
  return __builtin_amdgcn_rcpf(1.0f + __expf(-x));
}
__device__ __forceinline__ float tanh_fast(float x) {
  return 2.0f * __builtin_amdgcn_rcpf(1.0f + __expf(-2.0f * x)) - 1.0f;
}

// 16-bit A 16x32 fragment, contiguous k order used here: lane m=l&15, half h=l>>4,
// element i -> K = h*16+i
__device__ __forceinline__ int a_k(int h, int i) {
  return h * 16 + i;
}

// ---------------------------------------------------------------- utility
__global__ void zero_hn_kernel(float* __restrict__ hn) {
  int i = blockIdx.x * blockDim.x + threadIdx.x;
  if (i < B_ * N_ * HID_) hn[i] = 0.0f;
}

__global__ __launch_bounds__(256)
void edge_stats_kernel(const float* __restrict__ ea, float* __restrict__ stats) {
  __shared__ float r0[256], r1[256], r2[256], r3[256];
  int tid = threadIdx.x;
  float s0 = 0.f, s1 = 0.f, q0 = 0.f, q1 = 0.f;
  for (int e = tid; e < E_; e += 256) {
    float a = ea[e * 2 + 0], c = ea[e * 2 + 1];
    s0 += a; s1 += c; q0 += a * a; q1 += c * c;
  }
  r0[tid] = s0; r1[tid] = s1; r2[tid] = q0; r3[tid] = q1;
  __syncthreads();
  for (int o = 128; o > 0; o >>= 1) {
    if (tid < o) { r0[tid] += r0[tid+o]; r1[tid] += r1[tid+o];
                   r2[tid] += r2[tid+o]; r3[tid] += r3[tid+o]; }
    __syncthreads();
  }
  if (tid == 0) {
    float m0 = r0[0] / (float)E_, m1 = r1[0] / (float)E_;
    float v0 = (r2[0] - (float)E_ * m0 * m0) / (float)(E_ - 1);
    float v1 = (r3[0] - (float)E_ * m1 * m1) / (float)(E_ - 1);
    stats[0] = m0; stats[1] = m1; stats[2] = sqrtf(v0); stats[3] = sqrtf(v1);
  }
}

// Pack weights fragment-major: frag f, lane l, element i -> dst[f*512 + l*16 + i].
// B-fragment (contiguous k, as a_k): K = chunk*32 + (l>>4)*16 + i ; col = ntile*16 + (l&15).
// W_ih K-rows are permuted to match xc layout [x(32) | gnn(9) | pad(23)].
__global__ void convert_weights_kernel(const float* __restrict__ We1, const float* __restrict__ We2,
                                       const float* __restrict__ Wn,  const float* __restrict__ Wih,
                                       const float* __restrict__ Whh,
                                       f16* __restrict__ We1f, f16* __restrict__ We2f,
                                       f16* __restrict__ Wnf,  f16* __restrict__ Wihf,
                                       f16* __restrict__ Whhf) {
  const int T1 = 6 * 512, T2 = T1 + 2 * 512, T3 = T2 + 1 * 512,
            T4 = T3 + 24 * 512, T5 = T4 + 24 * 512;
  int idx = blockIdx.x * blockDim.x + threadIdx.x;
  if (idx >= T5) return;
  int rI = idx, kind;
  f16* dst;
  if      (idx < T1) {               dst = We1f; kind = 0; }
  else if (idx < T2) { rI -= T1;     dst = We2f; kind = 1; }
  else if (idx < T3) { rI -= T2;     dst = Wnf;  kind = 2; }
  else if (idx < T4) { rI -= T3;     dst = Wihf; kind = 3; }
  else               { rI -= T4;     dst = Whhf; kind = 4; }
  int f = rI >> 9, r = rI & 511;
  int lane = r >> 4, i = r & 15;
  int h = lane >> 4, nl = lane & 15;
  float v = 0.0f;
  if (kind == 0) {               // We1 67x32, f = c*2+nt
    int c = f >> 1, nt = f & 1;
    int k = c * 32 + h * 16 + i, col = nt * 16 + nl;
    if (k < 67) v = We1[k * 32 + col];
  } else if (kind == 1) {        // We2 32x30 (pad N), f = nt
    int k = h * 16 + i, col = f * 16 + nl;
    if (col < 30) v = We2[k * 30 + col];
  } else if (kind == 2) {        // Wn 30x9 (pad K,N)
    int k = h * 16 + i, col = nl;
    if (k < 30 && col < 9) v = Wn[k * 9 + col];
  } else if (kind == 3) {        // W_ih 41x192, K-permuted, f = j*2+c
    int j = f >> 1, c = f & 1;
    int k = c * 32 + h * 16 + i, col = j * 16 + nl;
    if (k < 32)      v = Wih[(9 + k) * 192 + col];   // x dims (orig rows 9..40)
    else if (k < 41) v = Wih[(k - 32) * 192 + col];  // gnn dims (orig rows 0..8)
  } else {                       // W_hh 64x192, f = j*2+c
    int j = f >> 1, c = f & 1;
    int k = c * 32 + h * 16 + i, col = j * 16 + nl;
    v = Whh[k * 192 + col];
  }
  dst[rI] = (f16)v;
}

__global__ void build_x_kernel(const float* __restrict__ rain, const float* __restrict__ feat,
                               const float* __restrict__ xnb, float* __restrict__ x,
                               float* __restrict__ agg, int t) {
  int tid = blockIdx.x * blockDim.x + threadIdx.x;
  if (tid >= B_ * N_) return;
  int b = tid / N_, n = tid % N_;
  float xn = (t == 0) ? rain[((size_t)b * HIST_ + (HIST_ - 1)) * N_ + n] : xnb[tid];
  float* xo = x + (size_t)tid * IN_DIM_;
  xo[0] = xn;
  const float* fr = feat + (((size_t)b * (HIST_ + PRED_) + (HIST_ + t)) * N_ + n) * FEAT_;
  #pragma unroll
  for (int k = 0; k < FEAT_; ++k) xo[1 + k] = fr[k];
  float* ag = agg + (size_t)tid * E_OUT_;
  #pragma unroll
  for (int k = 0; k < E_OUT_; ++k) ag[k] = 0.0f;
}

// ---------------------------------------------------------------- edge MLP
__global__ __launch_bounds__(256)
void edge_mlp_kernel(const float* __restrict__ x, const int* __restrict__ eidx,
                     const float* __restrict__ eattr,
                     const float* __restrict__ wmean, const float* __restrict__ wstd,
                     const f16* __restrict__ We1f, const float* __restrict__ be1,
                     const f16* __restrict__ We2f, const float* __restrict__ be2,
                     const float* __restrict__ stats, float* __restrict__ agg) {
  __shared__ f16 sE[8][16 * K1_PAD];
  __shared__ int sI[8][2][16];
  const int lane = threadIdx.x & 31;
  const int wv   = threadIdx.x >> 5;
  const int b    = blockIdx.y;
  const int m    = lane & 15;
  const int h    = lane >> 4;
  const float* xb = x + (size_t)b * N_ * IN_DIM_;

  __builtin_prefetch(We1f, 0, 0);     // global_prefetch_b8 (weights are L2-hot)

  // hoisted weight fragments: one v16h (2 x b128) per fragment per lane
  const v16h* W1v = (const v16h*)We1f;
  const v16h* W2v = (const v16h*)We2f;
  v16h bw1[3][2], bw2[2];
  #pragma unroll
  for (int c = 0; c < 3; ++c) {
    bw1[c][0] = W1v[(c * 2 + 0) * 32 + lane];
    bw1[c][1] = W1v[(c * 2 + 1) * 32 + lane];
  }
  bw2[0] = W2v[lane];
  bw2[1] = W2v[32 + lane];

  const int tile0 = (blockIdx.x * 8 + wv) * EDGE_TPW;
  for (int it = 0; it < EDGE_TPW; ++it) {
    const int e   = (tile0 + it) * 16 + m;
    const int src = eidx[e];
    const int tgt = eidx[E_ + e];

    if (h == 0) {
      const f4v* xs4 = (const f4v*)(xb + (size_t)src * IN_DIM_);
      #pragma unroll
      for (int q = 0; q < 8; ++q) {
        f4v v = xs4[q];
        h4 o; o[0] = (f16)v[0]; o[1] = (f16)v[1]; o[2] = (f16)v[2]; o[3] = (f16)v[3];
        *(h4*)&sE[wv][m * K1_PAD + q * 4] = o;
      }
      sI[wv][0][m] = src;
      sI[wv][1][m] = tgt;
    } else {
      const f4v* xt4 = (const f4v*)(xb + (size_t)tgt * IN_DIM_);
      #pragma unroll
      for (int q = 0; q < 8; ++q) {
        f4v v = xt4[q];
        h4 o; o[0] = (f16)v[0]; o[1] = (f16)v[1]; o[2] = (f16)v[2]; o[3] = (f16)v[3];
        *(h4*)&sE[wv][m * K1_PAD + 32 + q * 4] = o;
      }
      float dist = eattr[e * 2 + 0];
      float cdir = eattr[e * 2 + 1];
      float sp = xb[(size_t)src * IN_DIM_ + 30] * wstd[0] + wmean[0];
      float dr = xb[(size_t)src * IN_DIM_ + 31] * wstd[1] + wmean[1];
      float th = fabsf(cdir - dr);
      float ew = fmaxf(0.0f, 3.0f * sp * __cosf(th * 22.5f) * __builtin_amdgcn_rcpf(dist));
      h4 o; o[0] = (f16)((dist - stats[0]) / stats[2]);
            o[1] = (f16)((cdir - stats[1]) / stats[3]);
            o[2] = (f16)ew; o[3] = (f16)0.0f;
      *(h4*)&sE[wv][m * K1_PAD + 64] = o;
      h4 z = {};
      #pragma unroll
      for (int q = 17; q < 24; ++q) *(h4*)&sE[wv][m * K1_PAD + q * 4] = z;
    }
    __syncthreads();

    // layer 1: 16x96 @ 96x32, 6 WMMAs with chained accumulators
    v8f acc0 = {}, acc1 = {};
    #pragma unroll
    for (int c = 0; c < 3; ++c) {
      v16h a;
      #pragma unroll
      for (int i = 0; i < 16; ++i) a[i] = sE[wv][m * K1_PAD + c * 32 + a_k(h, i)];
      acc0 = __builtin_amdgcn_wmma_f32_16x16x32_f16(false, a, false, bw1[c][0], (short)0, acc0, false, false);
      acc1 = __builtin_amdgcn_wmma_f32_16x16x32_f16(false, a, false, bw1[c][1], (short)0, acc1, false, false);
    }
    __syncthreads();

    // sigmoid+bias, re-stage as 16x32 f16 A-tile for layer 2
    #pragma unroll
    for (int r = 0; r < 8; ++r) {
      int row = r + 8 * h;
      sE[wv][row * E_H_ + m]      = (f16)sigm(acc0[r] + be1[m]);
      sE[wv][row * E_H_ + 16 + m] = (f16)sigm(acc1[r] + be1[16 + m]);
    }
    __syncthreads();

    // layer 2: 16x32 @ 32x32 (30 valid cols)
    v16h a2;
    #pragma unroll
    for (int i = 0; i < 16; ++i) a2[i] = sE[wv][m * E_H_ + a_k(h, i)];
    v8f e0a = {}, e1a = {};
    e0a = __builtin_amdgcn_wmma_f32_16x16x32_f16(false, a2, false, bw2[0], (short)0, e0a, false, false);
    e1a = __builtin_amdgcn_wmma_f32_16x16x32_f16(false, a2, false, bw2[1], (short)0, e1a, false, false);

    // scatter: agg[tgt] += e ; agg[src] -= e
    #pragma unroll
    for (int r = 0; r < 8; ++r) {
      int row = r + 8 * h;
      int s  = sI[wv][0][row];
      int tg = sI[wv][1][row];
      {
        float v = sigm(e0a[r] + be2[m]);       // col = m (< 16 < 30)
        atomicAdd(&agg[((size_t)b * N_ + tg) * E_OUT_ + m],  v);
        atomicAdd(&agg[((size_t)b * N_ + s ) * E_OUT_ + m], -v);
      }
      if (m < E_OUT_ - 16) {                    // col = 16+m (< 30)
        float v = sigm(e1a[r] + be2[16 + m]);
        atomicAdd(&agg[((size_t)b * N_ + tg) * E_OUT_ + 16 + m],  v);
        atomicAdd(&agg[((size_t)b * N_ + s ) * E_OUT_ + 16 + m], -v);
      }
    }
    __syncthreads();   // staging reused next iteration
  }
}

// ---------------------------------------------------------------- node GNN head + GRU
__global__ __launch_bounds__(32)
void node_gru_kernel(const float* __restrict__ x, const float* __restrict__ agg,
                     float* __restrict__ hn,
                     const f16* __restrict__ Wnf,  const float* __restrict__ bn,
                     const f16* __restrict__ Wihf, const float* __restrict__ bih,
                     const f16* __restrict__ Whhf, const float* __restrict__ bhh,
                     const float* __restrict__ Wout, const float* __restrict__ bout,
                     float* __restrict__ xnb, float* __restrict__ out, int t) {
  __shared__ f16   sXC[16 * KXC_PAD];  // xc layout: [x(32) | gnn(9) | pad(23)]
  __shared__ f16   sH [16 * HID_];
  __shared__ f16   sA [16 * 32];
  __shared__ float sR [16 * HID_];
  __shared__ float sZ [16 * HID_];
  __shared__ float sHn[16 * HID_];
  const int lane = threadIdx.x;
  const int m = lane & 15, h = lane >> 4;
  const int b = blockIdx.y;
  const int n0 = blockIdx.x * 16;      // N_ = 625*16 exact
  const int node = n0 + m;

  // stage agg (pad 30->32)
  #pragma unroll
  for (int i = 0; i < 16; ++i) {
    int k = h * 16 + i;
    float v = (k < E_OUT_) ? agg[((size_t)b * N_ + node) * E_OUT_ + k] : 0.0f;
    sA[m * 32 + k] = (f16)v;
  }
  // stage x into xc cols 0..31 (vectorized), zero pad 41..63
  {
    const f4v* xr4 = (const f4v*)(x + ((size_t)b * N_ + node) * IN_DIM_);
    #pragma unroll
    for (int q = 0; q < 4; ++q) {
      f4v v = xr4[h * 4 + q];
      h4 o; o[0] = (f16)v[0]; o[1] = (f16)v[1]; o[2] = (f16)v[2]; o[3] = (f16)v[3];
      *(h4*)&sXC[m * KXC_PAD + h * 16 + q * 4] = o;
    }
  }
  if (h == 0) {
    #pragma unroll
    for (int i = 0; i < 12; ++i) sXC[m * KXC_PAD + 41 + i] = (f16)0.0f;
  } else {
    #pragma unroll
    for (int i = 0; i < 11; ++i) sXC[m * KXC_PAD + 53 + i] = (f16)0.0f;
  }
  // stage h (f32 -> f16, vectorized)
  {
    const f4v* hr4 = (const f4v*)(hn + ((size_t)b * N_ + node) * HID_);
    #pragma unroll
    for (int q = 0; q < 8; ++q) {
      f4v v = hr4[h * 8 + q];
      h4 o; o[0] = (f16)v[0]; o[1] = (f16)v[1]; o[2] = (f16)v[2]; o[3] = (f16)v[3];
      *(h4*)&sH[m * HID_ + h * 32 + q * 4] = o;
    }
  }
  __syncthreads();

  // GNN head: sigmoid(agg @ Wn + bn), one WMMA, write xc cols 32..40
  v16h aA, bWn = ((const v16h*)Wnf)[lane];
  #pragma unroll
  for (int i = 0; i < 16; ++i) aA[i] = sA[m * 32 + a_k(h, i)];
  v8f gacc = {};
  gacc = __builtin_amdgcn_wmma_f32_16x16x32_f16(false, aA, false, bWn, (short)0, gacc, false, false);
  if (m < GNN_OUT_) {
    float bb = bn[m];
    #pragma unroll
    for (int r = 0; r < 8; ++r)
      sXC[(r + 8 * h) * KXC_PAD + 32 + m] = (f16)sigm(gacc[r] + bb);
  }
  __syncthreads();

  // hoisted A fragments: xc (2 K-chunks) and h (2 K-chunks)
  v16h xcA[2], hA[2];
  #pragma unroll
  for (int c = 0; c < 2; ++c) {
    #pragma unroll
    for (int i = 0; i < 16; ++i) xcA[c][i] = sXC[m * KXC_PAD + c * 32 + a_k(h, i)];
    #pragma unroll
    for (int i = 0; i < 16; ++i) hA[c][i]  = sH[m * HID_ + c * 32 + a_k(h, i)];
  }

  const v16h* Wiv = (const v16h*)Wihf;
  const v16h* Whv = (const v16h*)Whhf;

  // gate tiles 0..7 : r (cols 0..63) and z (cols 64..127)
  for (int j = 0; j < 8; ++j) {
    const int col = j * 16 + m;
    v16h bi0 = Wiv[(j * 2 + 0) * 32 + lane];
    v16h bi1 = Wiv[(j * 2 + 1) * 32 + lane];
    v16h bh0 = Whv[(j * 2 + 0) * 32 + lane];
    v16h bh1 = Whv[(j * 2 + 1) * 32 + lane];
    v8f gi = {}, gh = {};
    gi = __builtin_amdgcn_wmma_f32_16x16x32_f16(false, xcA[0], false, bi0, (short)0, gi, false, false);
    gi = __builtin_amdgcn_wmma_f32_16x16x32_f16(false, xcA[1], false, bi1, (short)0, gi, false, false);
    gh = __builtin_amdgcn_wmma_f32_16x16x32_f16(false, hA[0],  false, bh0, (short)0, gh, false, false);
    gh = __builtin_amdgcn_wmma_f32_16x16x32_f16(false, hA[1],  false, bh1, (short)0, gh, false, false);
    float bgi = bih[col], bgh = bhh[col];
    #pragma unroll
    for (int r = 0; r < 8; ++r) {
      int row = r + 8 * h;
      float v = sigm((gi[r] + bgi) + (gh[r] + bgh));
      if (col < HID_) sR[row * HID_ + col] = v;
      else            sZ[row * HID_ + (col - HID_)] = v;
    }
  }
  __syncthreads();

  // gate tiles 8..11 : n = tanh(ig + r*hg); h' = (1-z)*n + z*h
  for (int j = 8; j < 12; ++j) {
    const int col = j * 16 + m;
    const int k = col - 128;
    v16h bi0 = Wiv[(j * 2 + 0) * 32 + lane];
    v16h bi1 = Wiv[(j * 2 + 1) * 32 + lane];
    v16h bh0 = Whv[(j * 2 + 0) * 32 + lane];
    v16h bh1 = Whv[(j * 2 + 1) * 32 + lane];
    v8f gi = {}, gh = {};
    gi = __builtin_amdgcn_wmma_f32_16x16x32_f16(false, xcA[0], false, bi0, (short)0, gi, false, false);
    gi = __builtin_amdgcn_wmma_f32_16x16x32_f16(false, xcA[1], false, bi1, (short)0, gi, false, false);
    gh = __builtin_amdgcn_wmma_f32_16x16x32_f16(false, hA[0],  false, bh0, (short)0, gh, false, false);
    gh = __builtin_amdgcn_wmma_f32_16x16x32_f16(false, hA[1],  false, bh1, (short)0, gh, false, false);
    float bgi = bih[col], bgh = bhh[col];
    #pragma unroll
    for (int r = 0; r < 8; ++r) {
      int row = r + 8 * h;
      float ig = gi[r] + bgi;
      float hg = gh[r] + bgh;
      float rv = sR[row * HID_ + k];
      float zv = sZ[row * HID_ + k];
      float nv = tanh_fast(ig + rv * hg);
      size_t hoff = ((size_t)b * N_ + n0 + row) * HID_ + k;
      float hold = hn[hoff];
      float hnew = (1.0f - zv) * nv + zv * hold;
      hn[hoff] = hnew;
      sHn[row * HID_ + k] = hnew;
    }
  }
  __syncthreads();

  // readout: xn = h' @ W_out + b_out
  if (h == 0) {
    float s = bout[0];
    #pragma unroll
    for (int k = 0; k < HID_; ++k) s += sHn[m * HID_ + k] * Wout[k];
    xnb[(size_t)b * N_ + node] = s;
    out[((size_t)b * PRED_ + t) * N_ + node] = s;
  }
}

// ---------------------------------------------------------------- launch
extern "C" void kernel_launch(void* const* d_in, const int* in_sizes, int n_in,
                              void* d_out, int out_size, void* d_ws, size_t ws_size,
                              hipStream_t stream) {
  (void)in_sizes; (void)n_in; (void)out_size; (void)ws_size;
  const float* rain  = (const float*)d_in[0];
  const float* feat  = (const float*)d_in[1];
  const int*   eidx  = (const int*)  d_in[2];
  const float* eattr = (const float*)d_in[3];
  const float* wmean = (const float*)d_in[4];
  const float* wstd  = (const float*)d_in[5];
  const float* We1   = (const float*)d_in[6];
  const float* be1   = (const float*)d_in[7];
  const float* We2   = (const float*)d_in[8];
  const float* be2   = (const float*)d_in[9];
  const float* Wn    = (const float*)d_in[10];
  const float* bn    = (const float*)d_in[11];
  const float* Wih   = (const float*)d_in[12];
  const float* bih   = (const float*)d_in[13];
  const float* Whh   = (const float*)d_in[14];
  const float* bhh   = (const float*)d_in[15];
  const float* Wout  = (const float*)d_in[16];
  const float* bout  = (const float*)d_in[17];
  float* out = (float*)d_out;

  char* ws = (char*)d_ws;
  size_t off = 0;
  auto alloc = [&](size_t bytes) -> char* {
    char* p = ws + off;
    off += (bytes + 255) & ~(size_t)255;
    return p;
  };
  float* x     = (float*)alloc((size_t)B_ * N_ * IN_DIM_ * 4);
  float* hn    = (float*)alloc((size_t)B_ * N_ * HID_ * 4);
  float* agg   = (float*)alloc((size_t)B_ * N_ * E_OUT_ * 4);
  float* xnb   = (float*)alloc((size_t)B_ * N_ * 4);
  float* stats = (float*)alloc(4 * 4);
  f16* We1f = (f16*)alloc(6  * 512 * 2);
  f16* We2f = (f16*)alloc(2  * 512 * 2);
  f16* Wnf  = (f16*)alloc(1  * 512 * 2);
  f16* Wihf = (f16*)alloc(24 * 512 * 2);
  f16* Whhf = (f16*)alloc(24 * 512 * 2);

  zero_hn_kernel<<<(B_ * N_ * HID_ + 255) / 256, 256, 0, stream>>>(hn);
  edge_stats_kernel<<<1, 256, 0, stream>>>(eattr, stats);
  const int wtot = (6 + 2 + 1 + 24 + 24) * 512;
  convert_weights_kernel<<<(wtot + 255) / 256, 256, 0, stream>>>(
      We1, We2, Wn, Wih, Whh, We1f, We2f, Wnf, Wihf, Whhf);

  for (int t = 0; t < PRED_; ++t) {
    build_x_kernel<<<(B_ * N_ + 255) / 256, 256, 0, stream>>>(rain, feat, xnb, x, agg, t);
    edge_mlp_kernel<<<dim3(E_ / (16 * 8 * EDGE_TPW), B_), 256, 0, stream>>>(
        x, eidx, eattr, wmean, wstd, We1f, be1, We2f, be2, stats, agg);
    node_gru_kernel<<<dim3(N_ / 16, B_), 32, 0, stream>>>(
        x, agg, hn, Wnf, bn, Wihf, bih, Whhf, bhh, Wout, bout, xnb, out, t);
  }
}
